// STIM_86251533238550
// MI455X (gfx1250) — matrix operand order repaired and checked by reading, not compiled, not run on hardware
//
#include <hip/hip_runtime.h>
#include <hip/hip_bf16.h>
#include <math.h>

// ---------------------------------------------------------------------------
// Problem constants (match reference)
// ---------------------------------------------------------------------------
#define B_SZ   4
#define L_SZ   1024
#define DM_SZ  1024
#define DI_SZ  1024
#define DS_SZ  16
#define DR_SZ  64
#define DC_SZ  4
#define NH_SZ  4
#define DK_SZ  256            // DM / NH
#define NTOK   (B_SZ * L_SZ)  // 4096 rows in every big GEMM
#define XROW_SZ (DR_SZ + 2 * DS_SZ)   // 96

// ---------------------------------------------------------------------------
// Types: bf16 + WMMA fragment vectors (CDNA5 / gfx1250, wave32)
// ---------------------------------------------------------------------------
typedef __bf16 bf16_t;
typedef __attribute__((ext_vector_type(16))) __bf16        v16bf;
typedef __attribute__((ext_vector_type(8)))  float         v8f;
typedef __attribute__((ext_vector_type(4)))  unsigned int  v4u;
typedef __attribute__((ext_vector_type(8)))  int           v8i;
typedef __attribute__((ext_vector_type(4)))  int           v4i;

union Frag16 {              // 32 bytes: one 16-element bf16 fragment
    v16bf f;
    v4u   q[2];
};

static __device__ inline bf16_t f2bf(float f) { return (bf16_t)f; }

static __device__ inline v8f wmma_bf16(v16bf a, v16bf b, v8f c) {
    return __builtin_amdgcn_wmma_f32_16x16x32_bf16(
        false, a, false, b, (short)0, c, false, false);
}

// ---------------------------------------------------------------------------
// Tensor Data Mover: 2D bf16 tile  global -> LDS  (ISA ch.8 D# layout)
// ---------------------------------------------------------------------------
#if __has_builtin(__builtin_amdgcn_tensor_load_to_lds)
#define HAVE_TDM 1
#else
#define HAVE_TDM 0
#endif

static __device__ inline void tdm_load_tile_bf16(
    unsigned lds_off, const void* gaddr,
    unsigned tensor_d0, unsigned tensor_d1,
    unsigned long long stride0, unsigned tile_d0, unsigned tile_d1)
{
#if HAVE_TDM
    unsigned long long ga = (unsigned long long)(uintptr_t)gaddr;
    v4u g0;
    g0.x = 1u;                                    // count=1, load, user mode
    g0.y = lds_off;                               // D#.lds_addr
    g0.z = (unsigned)(ga & 0xFFFFFFFFu);          // global_addr[31:0]
    g0.w = (unsigned)((ga >> 32) & 0x01FFFFFFu)   // global_addr[56:32]
         | (2u << 30);                            // type = 2 (image)
    v8i g1;
    g1[0] = (int)(1u << 16);                      // wg_mask=0, data_size=1 (2B)
    g1[1] = (int)((tensor_d0 & 0xFFFFu) << 16);   // tensor_dim0[15:0]
    g1[2] = (int)(((tensor_d0 >> 16) & 0xFFFFu) | ((tensor_d1 & 0xFFFFu) << 16));
    g1[3] = (int)(((tensor_d1 >> 16) & 0xFFFFu) | ((tile_d0   & 0xFFFFu) << 16));
    g1[4] = (int)(tile_d1 & 0xFFFFu);             // tile_dim1, tile_dim2=0
    g1[5] = (int)(stride0 & 0xFFFFFFFFull);       // tensor_dim0_stride[31:0]
    g1[6] = (int)((stride0 >> 32) & 0xFFFFull);   // stride[47:32]
    g1[7] = 0;
    v4i z4 = {0, 0, 0, 0};
#if defined(__clang_major__) && __clang_major__ >= 23
    v8i z8 = {0, 0, 0, 0, 0, 0, 0, 0};
    __builtin_amdgcn_tensor_load_to_lds(g0, g1, z4, z4, z8, 0);
#else
    __builtin_amdgcn_tensor_load_to_lds(g0, g1, z4, z4, 0);
#endif
#else
    (void)lds_off; (void)gaddr; (void)tensor_d0; (void)tensor_d1;
    (void)stride0; (void)tile_d0; (void)tile_d1;
#endif
}

static __device__ inline unsigned lds_byte_off(const void* p) {
    // low 32 bits of the generic LDS-aperture address are the LDS byte offset
    return (unsigned)(uintptr_t)p;
}

// ---------------------------------------------------------------------------
// Kernel 0: elementwise f32 -> bf16 conversion
// ---------------------------------------------------------------------------
__global__ __launch_bounds__(256) void cvt_bf16_kernel(
    const float* __restrict__ in, bf16_t* __restrict__ out, size_t n)
{
    size_t i = (size_t)blockIdx.x * blockDim.x + threadIdx.x;
    if (i < n) out[i] = f2bf(in[i]);
}

// ---------------------------------------------------------------------------
// Kernel 1: tiled WMMA GEMM on bf16 inputs, TDM double-buffered.
//   C[M,N] = act(A[M,K] @ W[N,K]^T + bias)
//   block = 128 thr (4 waves), 64x64 tile, wave owns 32x32 (2x2 WMMA frags).
//   Tile k+1 streams into LDS buffer (k+1)&1 via tensor_load_to_lds while
//   WMMAs consume buffer k&1; s_wait_tensorcnt 2 drains the previous pair.
//   ACT: 0 identity, 1 softplus.  OUTBF: 1 -> store bf16, 0 -> store f32.
// ---------------------------------------------------------------------------
template <int ACT, int OUTBF>
__global__ __launch_bounds__(128) void gemm_wmma_kernel(
    const bf16_t* __restrict__ A, int lda,
    const bf16_t* __restrict__ W, int ldw,
    const float* __restrict__ bias,
    void* __restrict__ Cout, int ldc,
    int M, int N, int K)
{
    __shared__ __align__(16) bf16_t sA[2][64][32];   // rows m, cols k (8 KB)
    __shared__ __align__(16) bf16_t sW[2][64][32];   // rows n, cols k (8 KB)

    const int tid   = threadIdx.x;
    const int lane  = tid & 31;
    const int wave  = tid >> 5;
    const int rlane = lane & 15;
    const int half  = lane >> 4;

    const int m0 = blockIdx.y * 64;
    const int n0 = blockIdx.x * 64;
    const int wm = (wave >> 1) * 32;
    const int wn = (wave & 1)  * 32;

    const int nk = K >> 5;          // K is a multiple of 32 for all call sites
    v8f acc[2][2] = {};

#if HAVE_TDM
    if (wave == 0) {                // prime the pipeline: tile 0 -> buffer 0
        tdm_load_tile_bf16(lds_byte_off(&sA[0][0][0]),
                           &A[(size_t)m0 * lda],
                           (unsigned)K, (unsigned)(M - m0),
                           (unsigned long long)lda, 32u, 64u);
        tdm_load_tile_bf16(lds_byte_off(&sW[0][0][0]),
                           &W[(size_t)n0 * ldw],
                           (unsigned)K, (unsigned)(N - n0),
                           (unsigned long long)ldw, 32u, 64u);
    }
#endif

    for (int kt = 0; kt < nk; ++kt) {
        const int cur = kt & 1;
#if HAVE_TDM
        if (wave == 0) {
            if (kt + 1 < nk) {      // stream next tile into the other buffer
                const int nxt = (kt + 1) & 1;
                const int k1 = (kt + 1) << 5;
                tdm_load_tile_bf16(lds_byte_off(&sA[nxt][0][0]),
                                   &A[(size_t)m0 * lda + k1],
                                   (unsigned)(K - k1), (unsigned)(M - m0),
                                   (unsigned long long)lda, 32u, 64u);
                tdm_load_tile_bf16(lds_byte_off(&sW[nxt][0][0]),
                                   &W[(size_t)n0 * ldw + k1],
                                   (unsigned)(K - k1), (unsigned)(N - n0),
                                   (unsigned long long)ldw, 32u, 64u);
                // in-order completion: <=2 outstanding => tile kt is resident
                __builtin_amdgcn_s_wait_tensorcnt(2);
            } else {
                __builtin_amdgcn_s_wait_tensorcnt(0);
            }
        }
#else
        {
            const int k0 = kt << 5;
            for (int i = tid; i < 64 * 32; i += 128) {
                int r = i >> 5, c = i & 31;
                sA[cur][r][c] = ((m0 + r) < M && (k0 + c) < K)
                    ? A[(size_t)(m0 + r) * lda + (k0 + c)] : (bf16_t)0.f;
                sW[cur][r][c] = ((n0 + r) < N && (k0 + c) < K)
                    ? W[(size_t)(n0 + r) * ldw + (k0 + c)] : (bf16_t)0.f;
            }
        }
#endif
        __syncthreads();

        // fragment assembly: pure 128-bit LDS loads, no unpacking VALU
        Frag16 af[2], bfm[2];
        #pragma unroll
        for (int mi = 0; mi < 2; ++mi) {
            const int row = wm + mi * 16 + rlane;
            af[mi].q[0] = *(const v4u*)&sA[cur][row][half * 8];       // K 0..15
            af[mi].q[1] = *(const v4u*)&sA[cur][row][16 + half * 8];  // K 16..31
        }
        #pragma unroll
        for (int ni = 0; ni < 2; ++ni) {
            const int col = wn + ni * 16 + rlane;
            bfm[ni].q[0] = *(const v4u*)&sW[cur][col][half * 8];
            bfm[ni].q[1] = *(const v4u*)&sW[cur][col][16 + half * 8];
        }
        #pragma unroll
        for (int mi = 0; mi < 2; ++mi)
            #pragma unroll
            for (int ni = 0; ni < 2; ++ni)
                acc[mi][ni] = wmma_bf16(af[mi].f, bfm[ni].f, acc[mi][ni]);

        __syncthreads();    // all reads of buffer `cur` done before it refills
    }

    // epilogue
    #pragma unroll
    for (int mi = 0; mi < 2; ++mi) {
        #pragma unroll
        for (int ni = 0; ni < 2; ++ni) {
            #pragma unroll
            for (int j = 0; j < 8; ++j) {
                const int gm = m0 + wm + mi * 16 + j + 8 * half;  // M = j + 8*half
                const int gn = n0 + wn + ni * 16 + rlane;         // N = lane%16
                if (gm < M && gn < N) {
                    float v = acc[mi][ni][j];
                    if (bias) v += bias[gn];
                    if (ACT == 1)
                        v = (v > 20.f) ? v : log1pf(__expf(v));
                    if (OUTBF)
                        ((bf16_t*)Cout)[(size_t)gm * ldc + gn] = f2bf(v);
                    else
                        ((float*)Cout)[(size_t)gm * ldc + gn] = v;
                }
            }
        }
    }
}

// ---------------------------------------------------------------------------
// Kernel 2: causal depthwise conv1d (k=4) + bias + SiLU -> u (f32 AND bf16)
// ---------------------------------------------------------------------------
__global__ __launch_bounds__(256) void conv_silu_kernel(
    const float* __restrict__ xx, const float* __restrict__ cw,
    const float* __restrict__ cb,
    float* __restrict__ u_f, bf16_t* __restrict__ u_b)
{
    const size_t idx = (size_t)blockIdx.x * blockDim.x + threadIdx.x;
    const size_t total = (size_t)B_SZ * L_SZ * DI_SZ;
    if (idx >= total) return;
    const int d = (int)(idx % DI_SZ);
    const int l = (int)((idx / DI_SZ) % L_SZ);
    const int b = (int)(idx / ((size_t)DI_SZ * L_SZ));

    float acc = cb[d];
    #pragma unroll
    for (int j = 0; j < DC_SZ; ++j) {
        const int ls = l - (DC_SZ - 1) + j;
        if (ls >= 0)
            acc += xx[((size_t)b * L_SZ + ls) * DI_SZ + d] * cw[d * DC_SZ + j];
    }
    const float s = acc / (1.0f + __expf(-acc));
    u_f[idx] = s;
    u_b[idx] = f2bf(s);
}

// ---------------------------------------------------------------------------
// Kernel 3: selective scan; one thread per (b, d), DS=16 state in registers,
// B/C vectors staged through LDS each timestep.  Emits y as bf16 (GEMM input).
// ---------------------------------------------------------------------------
__global__ __launch_bounds__(256) void scan_kernel(
    const float* __restrict__ delta, const float* __restrict__ xdbl,
    const float* __restrict__ u, const float* __restrict__ A_log,
    const float* __restrict__ Dv, bf16_t* __restrict__ y)
{
    const int b = blockIdx.y;
    const int d = blockIdx.x * blockDim.x + threadIdx.x;

    __shared__ float sBC[2 * DS_SZ];

    float Arow[DS_SZ];
    #pragma unroll
    for (int n = 0; n < DS_SZ; ++n)
        Arow[n] = -__expf(A_log[(size_t)d * DS_SZ + n]);
    float h[DS_SZ];
    #pragma unroll
    for (int n = 0; n < DS_SZ; ++n) h[n] = 0.f;
    const float Dd = Dv[d];

    for (int l = 0; l < L_SZ; ++l) {
        if (threadIdx.x < 2 * DS_SZ)
            sBC[threadIdx.x] =
                xdbl[((size_t)b * L_SZ + l) * XROW_SZ + DR_SZ + threadIdx.x];
        __syncthreads();

        const size_t ti = ((size_t)b * L_SZ + l) * DI_SZ + d;
        const float dv = delta[ti];
        const float uv = u[ti];
        float yv = 0.f;
        #pragma unroll
        for (int n = 0; n < DS_SZ; ++n) {
            const float dA = __expf(dv * Arow[n]);
            h[n] = dA * h[n] + dv * sBC[n] * uv;
            yv  += h[n] * sBC[DS_SZ + n];
        }
        y[ti] = f2bf(yv + uv * Dd);
        __syncthreads();
    }
}

// ---------------------------------------------------------------------------
// Kernel 4: transpose V (B,L,DM)bf16 -> Vt[(b*DM + c)*L + l] for contiguous
// key-dimension loads in attention phase 3.
// ---------------------------------------------------------------------------
__global__ __launch_bounds__(256) void transpose_v_kernel(
    const bf16_t* __restrict__ Vb, bf16_t* __restrict__ Vt)
{
    const size_t idx = (size_t)blockIdx.x * blockDim.x + threadIdx.x;
    const size_t total = (size_t)B_SZ * L_SZ * DM_SZ;
    if (idx >= total) return;
    const int c = (int)(idx % DM_SZ);
    const int l = (int)((idx / DM_SZ) % L_SZ);
    const int b = (int)(idx / ((size_t)DM_SZ * L_SZ));
    Vt[((size_t)b * DM_SZ + c) * L_SZ + l] = Vb[idx];
}

// ---------------------------------------------------------------------------
// Kernel 5: cross attention, WMMA flash-style, bf16 operands.
//   One workgroup (8 waves) per (b, h, 16-query tile).
// ---------------------------------------------------------------------------
__global__ __launch_bounds__(256) void attn_wmma_kernel(
    const bf16_t* __restrict__ Qb,   // (B, L, DM)
    const bf16_t* __restrict__ Kb,   // (B, L, DM)
    const bf16_t* __restrict__ Vt,   // (B, DM, L) transposed
    bf16_t* __restrict__ O)          // (B, L, DM) bf16 (feeds wo GEMM)
{
    __shared__ float sS[16][L_SZ];   // 16 x 1024 f32 = 64 KB scores

    const int q0    = blockIdx.x * 16;
    const int h     = blockIdx.y;
    const int b     = blockIdx.z;
    const int tid   = threadIdx.x;
    const int lane  = tid & 31;
    const int wave  = tid >> 5;
    const int rlane = lane & 15;
    const int half  = lane >> 4;

    const size_t base = (size_t)b * L_SZ * DM_SZ + (size_t)h * DK_SZ;

    // preload Q A-fragments for the 8 chunks of DK=256 (pure b128 loads)
    Frag16 qf[8];
    {
        const bf16_t* qrow = Qb + base + (size_t)(q0 + rlane) * DM_SZ;
        #pragma unroll
        for (int c = 0; c < 8; ++c) {
            qf[c].q[0] = *(const v4u*)(qrow + c * 32 + half * 8);
            qf[c].q[1] = *(const v4u*)(qrow + c * 32 + 16 + half * 8);
        }
    }

    // phase 1: S = Q K^T
    const float scale = 0.0625f;     // 1/sqrt(256)
    for (int kt = wave; kt < L_SZ / 16; kt += 8) {
        const int key0 = kt * 16;
        v8f s = {};
        const bf16_t* krow = Kb + base + (size_t)(key0 + rlane) * DM_SZ;
        #pragma unroll
        for (int c = 0; c < 8; ++c) {
            Frag16 kb;
            kb.q[0] = *(const v4u*)(krow + c * 32 + half * 8);
            kb.q[1] = *(const v4u*)(krow + c * 32 + 16 + half * 8);
            s = wmma_bf16(qf[c].f, kb.f, s);
        }
        #pragma unroll
        for (int j = 0; j < 8; ++j)
            sS[j + 8 * half][key0 + rlane] = s[j] * scale;
    }
    __syncthreads();

    // phase 2: softmax, each wave owns 2 rows
    for (int r = wave * 2; r < wave * 2 + 2; ++r) {
        float mx = -1e30f;
        for (int i = lane; i < L_SZ; i += 32) mx = fmaxf(mx, sS[r][i]);
        #pragma unroll
        for (int off = 16; off > 0; off >>= 1)
            mx = fmaxf(mx, __shfl_xor(mx, off, 32));
        float sum = 0.f;
        for (int i = lane; i < L_SZ; i += 32) {
            const float e = __expf(sS[r][i] - mx);
            sS[r][i] = e;
            sum += e;
        }
        #pragma unroll
        for (int off = 16; off > 0; off >>= 1)
            sum += __shfl_xor(sum, off, 32);
        const float inv = 1.0f / sum;
        for (int i = lane; i < L_SZ; i += 32) sS[r][i] *= inv;
    }
    __syncthreads();

    // phase 3: O = P V  (each wave owns 2 of 16 output column tiles)
    for (int ct = wave; ct < DK_SZ / 16; ct += 8) {
        const int dv0 = ct * 16;
        const bf16_t* vrow =
            Vt + ((size_t)b * DM_SZ + h * DK_SZ + dv0 + rlane) * L_SZ;
        v8f o = {};
        for (int kc = 0; kc < L_SZ / 32; ++kc) {
            const int kbase = kc * 32;
            Frag16 pa, vb;
            #pragma unroll
            for (int e = 0; e < 8; ++e) {
                pa.f[e]     = f2bf(sS[rlane][kbase + half * 8 + e]);
                pa.f[8 + e] = f2bf(sS[rlane][kbase + 16 + half * 8 + e]);
            }
            vb.q[0] = *(const v4u*)(vrow + kbase + half * 8);
            vb.q[1] = *(const v4u*)(vrow + kbase + 16 + half * 8);
            o = wmma_bf16(pa.f, vb.f, o);
        }
        #pragma unroll
        for (int j = 0; j < 8; ++j) {
            const int m = q0 + j + 8 * half;
            O[base + (size_t)m * DM_SZ + dv0 + rlane] = f2bf(o[j]);
        }
    }
}

// ---------------------------------------------------------------------------
// Host-side launch
// ---------------------------------------------------------------------------
static inline unsigned ceil_div(size_t a, unsigned b) {
    return (unsigned)((a + b - 1) / b);
}

extern "C" void kernel_launch(void* const* d_in, const int* in_sizes, int n_in,
                              void* d_out, int out_size, void* d_ws, size_t ws_size,
                              hipStream_t stream)
{
    (void)in_sizes; (void)n_in; (void)out_size; (void)ws_size;

    const float* x         = (const float*)d_in[0];
    const float* xx        = (const float*)d_in[1];
    const float* in_proj_w = (const float*)d_in[2];
    const float* conv_w    = (const float*)d_in[3];
    const float* conv_b    = (const float*)d_in[4];
    const float* x_proj_w  = (const float*)d_in[5];
    const float* dt_proj_w = (const float*)d_in[6];
    const float* dt_proj_b = (const float*)d_in[7];
    const float* A_log     = (const float*)d_in[8];
    const float* Dv        = (const float*)d_in[9];
    const float* wq        = (const float*)d_in[10];
    const float* bq        = (const float*)d_in[11];
    const float* wk        = (const float*)d_in[12];
    const float* bk        = (const float*)d_in[13];
    const float* wv        = (const float*)d_in[14];
    const float* bv        = (const float*)d_in[15];
    const float* wo        = (const float*)d_in[16];
    const float* bo        = (const float*)d_in[17];
    float* out = (float*)d_out;

    const size_t T  = (size_t)NTOK;            // 4096
    const size_t TD = T * DM_SZ;               // 4M elements

    // --- workspace carve-out ---
    char* wsb = (char*)d_ws;
    auto take_f32 = [&](size_t n) { float*  p = (float*) wsb; wsb += n * 4; return p; };
    auto take_bf  = [&](size_t n) { bf16_t* p = (bf16_t*)wsb; wsb += ((n * 2 + 15) & ~15ull); return p; };

    float*  u_f    = take_f32(TD);
    float*  xdbl   = take_f32(T * XROW_SZ);
    float*  delta  = take_f32(TD);

    bf16_t* x_bf   = take_bf(TD);
    bf16_t* q_bf   = take_bf(TD);
    bf16_t* u_bf   = take_bf(TD);
    bf16_t* xdbl_b = take_bf(T * XROW_SZ);
    bf16_t* y_bf   = take_bf(TD);
    bf16_t* Qb     = take_bf(TD);
    bf16_t* Kb     = take_bf(TD);
    bf16_t* Vb     = take_bf(TD);
    bf16_t* Vt     = take_bf(TD);
    bf16_t* ao     = take_bf(TD);

    bf16_t* inproj_b = take_bf((size_t)DI_SZ * DM_SZ);
    bf16_t* xproj_b  = take_bf((size_t)XROW_SZ * DI_SZ);
    bf16_t* dtproj_b = take_bf((size_t)DI_SZ * DR_SZ);
    bf16_t* wq_b     = take_bf((size_t)DM_SZ * DM_SZ);
    bf16_t* wk_b     = take_bf((size_t)DM_SZ * DM_SZ);
    bf16_t* wv_b     = take_bf((size_t)DM_SZ * DM_SZ);
    bf16_t* wo_b     = take_bf((size_t)DM_SZ * DM_SZ);

    const dim3 blk256(256), gemm_blk(128);
    const int M = (int)T;

    auto cvt = [&](const float* src, bf16_t* dst, size_t n) {
        cvt_bf16_kernel<<<dim3(ceil_div(n, 256)), blk256, 0, stream>>>(src, dst, n);
    };

    // 0) one-time bf16 conversions of activations-in and all weights
    cvt(x, x_bf, TD);
    cvt(in_proj_w, inproj_b, (size_t)DI_SZ * DM_SZ);
    cvt(x_proj_w,  xproj_b,  (size_t)XROW_SZ * DI_SZ);
    cvt(dt_proj_w, dtproj_b, (size_t)DI_SZ * DR_SZ);
    cvt(wq, wq_b, (size_t)DM_SZ * DM_SZ);
    cvt(wk, wk_b, (size_t)DM_SZ * DM_SZ);
    cvt(wv, wv_b, (size_t)DM_SZ * DM_SZ);
    cvt(wo, wo_b, (size_t)DM_SZ * DM_SZ);

    // 1) q = x @ in_proj_w^T  -> bf16
    {
        dim3 grid(DI_SZ / 64, (M + 63) / 64);
        gemm_wmma_kernel<0, 1><<<grid, gemm_blk, 0, stream>>>(
            x_bf, DM_SZ, inproj_b, DM_SZ, nullptr, q_bf, DI_SZ, M, DI_SZ, DM_SZ);
    }
    // 2) u = silu(causal dwconv(xx) + conv_b)  (f32 + bf16)
    conv_silu_kernel<<<dim3(ceil_div(TD, 256)), blk256, 0, stream>>>(
        xx, conv_w, conv_b, u_f, u_bf);
    // 3) x_dbl = u @ x_proj_w^T  (f32; 96 cols)
    {
        dim3 grid((XROW_SZ + 63) / 64, (M + 63) / 64);
        gemm_wmma_kernel<0, 0><<<grid, gemm_blk, 0, stream>>>(
            u_bf, DI_SZ, xproj_b, DI_SZ, nullptr, xdbl, XROW_SZ, M, XROW_SZ, DI_SZ);
    }
    cvt(xdbl, xdbl_b, T * XROW_SZ);
    // 4) delta = softplus(x_dbl[:, :64] @ dt_proj_w^T + dt_proj_b)  (f32)
    {
        dim3 grid(DI_SZ / 64, (M + 63) / 64);
        gemm_wmma_kernel<1, 0><<<grid, gemm_blk, 0, stream>>>(
            xdbl_b, XROW_SZ, dtproj_b, DR_SZ, dt_proj_b, delta, DI_SZ, M, DI_SZ, DR_SZ);
    }
    // 5) selective scan -> y (bf16)
    scan_kernel<<<dim3(DI_SZ / 256, B_SZ), blk256, 0, stream>>>(
        delta, xdbl, u_f, A_log, Dv, y_bf);
    // 6-8) Q/K/V projections -> bf16
    {
        dim3 grid(DM_SZ / 64, (M + 63) / 64);
        gemm_wmma_kernel<0, 1><<<grid, gemm_blk, 0, stream>>>(
            q_bf, DI_SZ, wq_b, DM_SZ, bq, Qb, DM_SZ, M, DM_SZ, DM_SZ);
        gemm_wmma_kernel<0, 1><<<grid, gemm_blk, 0, stream>>>(
            y_bf, DI_SZ, wk_b, DM_SZ, bk, Kb, DM_SZ, M, DM_SZ, DM_SZ);
        gemm_wmma_kernel<0, 1><<<grid, gemm_blk, 0, stream>>>(
            y_bf, DI_SZ, wv_b, DM_SZ, bv, Vb, DM_SZ, M, DM_SZ, DM_SZ);
    }
    // 9) V transpose for contiguous key-dim loads
    transpose_v_kernel<<<dim3(ceil_div(TD, 256)), blk256, 0, stream>>>(Vb, Vt);
    // 10) attention -> ao (bf16)
    attn_wmma_kernel<<<dim3(L_SZ / 16, NH_SZ, B_SZ), blk256, 0, stream>>>(
        Qb, Kb, Vt, ao);
    // 11) out = ao @ wo^T + bo  (f32 -> d_out)
    {
        dim3 grid(DM_SZ / 64, (M + 63) / 64);
        gemm_wmma_kernel<0, 0><<<grid, gemm_blk, 0, stream>>>(
            ao, DM_SZ, wo_b, DM_SZ, bo, out, DM_SZ, M, DM_SZ, DM_SZ);
    }
}
